// BasicMotionEncoder_54597624266890
// MI455X (gfx1250) — matrix operand order repaired and checked by reading, not compiled, not run on hardware
//
#include <hip/hip_runtime.h>
#include <hip/hip_bf16.h>
#include <stdint.h>

// ---------------------------------------------------------------------------
// BasicMotionEncoder on MI455X (gfx1250).
// All convs = implicit GEMM on v_wmma_f32_16x16x32_bf16.
// - activations bf16 end-to-end (f32 only at graph edges)
// - weights pre-padded to [Mpad][Kpad] bf16 -> A tiles DMA'd by the Tensor
//   Data Mover (tensor_load_to_lds) with hardware LDS row padding
// - LDS double buffered, one barrier per K-step, wave32 / 8-wave blocks
// ---------------------------------------------------------------------------

#ifndef __has_builtin
#define __has_builtin(x) 0
#endif

#if __has_builtin(__builtin_amdgcn_tensor_load_to_lds)
#define USE_TDM 1
#else
#define USE_TDM 0
#endif

typedef __bf16 v16bf __attribute__((ext_vector_type(16)));
typedef float  v8f   __attribute__((ext_vector_type(8)));
typedef unsigned int u32x4 __attribute__((ext_vector_type(4)));
typedef int          i32x4 __attribute__((ext_vector_type(4)));
typedef int          i32x8 __attribute__((ext_vector_type(8)));

#define Bc   8
#define Hc   96
#define Wc   128
#define HWc  12288          // 96*128
#define NTOT 98304          // 8*96*128

__device__ __forceinline__ unsigned short f2bf(float f) {
    unsigned u = __float_as_uint(f);
    u += 0x7FFFu + ((u >> 16) & 1u);      // round-to-nearest-even
    return (unsigned short)(u >> 16);
}

union ABOp {
    v16bf v;
    uint4 q[2];
};

#if USE_TDM
// Issue one TDM 2D tile load: 64 rows x 32 bf16 from padded weight matrix
// (row stride Kpad elements) into LDS with 16-short row padding -> 48-short
// LDS rows (96B, 16B aligned, bank-spread). Descriptor per CDNA5 D# layout.
__device__ __forceinline__ void tdm_issue_A(const unsigned short* gsrc,
                                            unsigned int lds_off,
                                            int Kpad, int Mpad) {
    unsigned long long ga = (unsigned long long)(uintptr_t)gsrc;
    u32x4 g0;
    g0[0] = 1u;                                    // count=1, user descriptor
    g0[1] = lds_off;                               // lds_addr (bytes)
    g0[2] = (unsigned int)ga;                      // global_addr[31:0]
    g0[3] = (unsigned int)((ga >> 32) & 0x01FFFFFFull) | (2u << 30); // [56:32] | type=2
    i32x8 g1;
    g1[0] = (1 << 16)      // data_size = 2 bytes
          | (1 << 20)      // pad_enable
          | (3 << 22)      // pad_interval: every 16 DWORDs (one 32-bf16 row)
          | (7 << 25);     // pad_amount: 8 DWORDs (16 bf16) -> 48-short rows
    g1[1] = (Kpad & 0xFFFF) << 16;                             // tensor_dim0 lo16
    g1[2] = ((unsigned)Kpad >> 16) | ((Mpad & 0xFFFF) << 16);  // dim0 hi | dim1 lo
    g1[3] = ((unsigned)Mpad >> 16) | (32u << 16);              // dim1 hi | tile_dim0=32
    g1[4] = 64;                                                // tile_dim1=64, tile_dim2=0
    g1[5] = Kpad;                                              // tensor_dim0_stride[31:0]
    g1[6] = 0;                                                 // stride hi | dim1_stride lo
    g1[7] = 0;
    i32x4 z4 = {0, 0, 0, 0};
#if __clang_major__ >= 23
    i32x8 z8 = {0, 0, 0, 0, 0, 0, 0, 0};
    __builtin_amdgcn_tensor_load_to_lds(g0, g1, z4, z4, z8, 0);
#else
    __builtin_amdgcn_tensor_load_to_lds(g0, g1, z4, z4, 0);
#endif
}
#endif

// Implicit GEMM conv: M = Cout (tile 64), N = spatial (tile 128, one full W
// row), K = Cin*KH*KW stepped by 32. x/weights bf16; output bf16 or f32.
template <int KH, int KW, int PAD, bool OUT_BF16>
__global__ __launch_bounds__(256) void conv_wmma(
    const unsigned short* __restrict__ x,     // bf16 activations, NCHW
    const unsigned short* __restrict__ wpad,  // bf16 weights [Mpad][Kpad], zero-padded
    const float* __restrict__ bias,
    void* __restrict__ y,
    int Cin, int Cout, int co_off, int co_tot, int Kpad, int Mpad) {

    constexpr int KHW = KH * KW;
    const int Ktot = Cin * KHW;
    const int NK   = Kpad >> 5;

    __shared__ __align__(16) unsigned short lA[2][64][48];
    __shared__ __align__(16) unsigned short lB[2][128][48];

    const int tid    = threadIdx.x;
    const int n_base = blockIdx.x * 128;            // one full output row
    const int m_base = blockIdx.y * 64;
    const int bb  = n_base / HWc;                   // block-uniform batch
    const int hh0 = (n_base - bb * HWc) >> 7;       // block-uniform row

    const int lane = tid & 31;
    const int wave = tid >> 5;
    const int wm   = wave & 3;     // 4 M sub-tiles of 16
    const int wn   = wave >> 2;    // 2 N sub-tiles of 64
    const int half = lane >> 4;
    const int lm   = lane & 15;

#if USE_TDM
    unsigned int ldsA_off[2];
    ldsA_off[0] = (unsigned int)(uintptr_t)&lA[0][0][0];
    ldsA_off[1] = (unsigned int)(uintptr_t)&lA[1][0][0];
#endif

    auto loadA = [&](int buf, int ks) {
#if USE_TDM
        if (wave == 0)
            tdm_issue_A(wpad + (size_t)m_base * Kpad + (size_t)ks * 32,
                        ldsA_off[buf], Kpad, Mpad);
#else
        const int kbase = ks * 32;
        #pragma unroll
        for (int p = 0; p < 8; ++p) {
            int idx = p * 256 + tid;
            int k = idx & 31, m = idx >> 5;
            lA[buf][m][k] = wpad[(size_t)(m_base + m) * Kpad + kbase + k];
        }
#endif
    };

    auto stageB = [&](int buf, int ks) {
        const int kbase = ks * 32;
        #pragma unroll
        for (int p = 0; p < 16; ++p) {
            int idx = p * 256 + tid;
            int n = idx & 127, k = idx >> 7;        // consecutive tid -> consecutive n
            int kg = kbase + k;
            unsigned short v = 0;
            if (kg < Ktot) {
                int ci = kg / KHW;                  // compile-time magic divide
                int r  = kg - ci * KHW;
                int kh = r / KW;
                int kw = r - kh * KW;
                int hh = hh0 + kh - PAD;
                int ww = n + kw - PAD;
                if (hh >= 0 && hh < Hc && (unsigned)ww < (unsigned)Wc)
                    v = x[((size_t)bb * Cin + ci) * HWc + hh * Wc + ww];
            }
            lB[buf][n][k] = v;
        }
    };

    loadA(0, 0);
    stageB(0, 0);

    // CDNA5 16-bit operand layouts -> two ds_load_b128 per operand
    const int am    = wm * 16 + lm;
    const int koffA = half ? 8  : 0;   // A: lanes>=16 start at K=8 / K=24
    const int nn    = wn * 64 + lm;
    const int koffB = half ? 8  : 0;   // B: lanes>=16 start at K=8 / K=24

    v8f acc[4] = {};

    for (int ks = 0; ks < NK; ++ks) {
#if USE_TDM
        if (wave == 0) __builtin_amdgcn_s_wait_tensorcnt(0);  // A tile ks landed
#endif
        __syncthreads();   // publish A+B tile ks; all waves done with buf (ks+1)&1
        if (ks + 1 < NK) {
            loadA((ks + 1) & 1, ks + 1);     // TDM overlaps with compute below
            stageB((ks + 1) & 1, ks + 1);
        }
        const int buf = ks & 1;
        ABOp a;
        a.q[0] = *(const uint4*)&lA[buf][am][koffA];
        a.q[1] = *(const uint4*)&lA[buf][am][koffA + 16];
        #pragma unroll
        for (int s = 0; s < 4; ++s) {
            ABOp b;
            b.q[0] = *(const uint4*)&lB[buf][nn + s * 16][koffB];
            b.q[1] = *(const uint4*)&lB[buf][nn + s * 16][koffB + 16];
            acc[s] = __builtin_amdgcn_wmma_f32_16x16x32_bf16(
                         false, a.v, false, b.v, (short)0, acc[s], false, false);
        }
    }

    // epilogue: bias + ReLU, write bf16 (intermediate) or f32 (final)
    // C layout: VGPR j -> M=j (lanes 0-15) / M=j+8 (lanes 16-31); N=lane%16
    const int cobase = m_base + wm * 16 + half * 8;
    const size_t obase = ((size_t)bb * co_tot + co_off) * HWc + (size_t)hh0 * Wc;
    #pragma unroll
    for (int j = 0; j < 8; ++j) {
        int co = cobase + j;
        if (co < Cout) {
            float bv = bias[co];
            #pragma unroll
            for (int s = 0; s < 4; ++s) {
                float v = acc[s][j] + bv;
                v = v > 0.f ? v : 0.f;
                size_t off = obase + (size_t)co * HWc + wn * 64 + s * 16 + lm;
                if (OUT_BF16) ((unsigned short*)y)[off] = f2bf(v);
                else          ((float*)y)[off] = v;
            }
        }
    }
}

// f32 -> bf16 elementwise (graph-edge inputs)
__global__ void cvt_bf16(const float* __restrict__ src,
                         unsigned short* __restrict__ dst, int n) {
    int i = blockIdx.x * 256 + threadIdx.x;
    if (i < n) dst[i] = f2bf(src[i]);
}

// f32 OIHW weights -> bf16 [Mpad][Kpad], zero-padded (enables pure TDM tiles)
__global__ void cvt_wpad(const float* __restrict__ src,
                         unsigned short* __restrict__ dst,
                         int Cout, int Ktot, int Kpad, int total) {
    int i = blockIdx.x * 256 + threadIdx.x;
    if (i >= total) return;
    int m = i / Kpad, k = i - m * Kpad;
    unsigned short v = 0;
    if (m < Cout && k < Ktot) v = f2bf(src[(size_t)m * Ktot + k]);
    dst[i] = v;
}

// flow (f32 input) -> output channels 126..127 (final concat)
__global__ void copy_flow(const float* __restrict__ flow, float* __restrict__ out) {
    int i = blockIdx.x * 256 + threadIdx.x;
    if (i >= Bc * 2 * HWc) return;
    int b = i / (2 * HWc);
    int r = i - b * (2 * HWc);
    int c = r / HWc;
    int p = r - c * HWc;
    out[((size_t)b * 128 + 126 + c) * HWc + p] = flow[i];
}

extern "C" void kernel_launch(void* const* d_in, const int* in_sizes, int n_in,
                              void* d_out, int out_size, void* d_ws, size_t ws_size,
                              hipStream_t stream) {
    const float* flow = (const float*)d_in[0];
    const float* corr = (const float*)d_in[1];
    const float* wc1  = (const float*)d_in[2];
    const float* bc1  = (const float*)d_in[3];
    const float* wc2  = (const float*)d_in[4];
    const float* bc2  = (const float*)d_in[5];
    const float* wf1  = (const float*)d_in[6];
    const float* bf1  = (const float*)d_in[7];
    const float* wf2  = (const float*)d_in[8];
    const float* bf2  = (const float*)d_in[9];
    const float* wo   = (const float*)d_in[10];
    const float* bo   = (const float*)d_in[11];
    float* out = (float*)d_out;

    // workspace layout (all bf16 raw u16); ~167 MB total
    typedef unsigned short u16;
    u16* corr_bf = (u16*)d_ws;                          // [8,324,HW]
    u16* t1_bf   = corr_bf + (size_t)Bc * 324 * HWc;    // [8,256,HW] (reused as flo1)
    u16* cat_bf  = t1_bf   + (size_t)Bc * 256 * HWc;    // [8,256,HW]
    u16* flow_bf = cat_bf  + (size_t)Bc * 256 * HWc;    // [8,2,HW]
    u16* w1p  = flow_bf + (size_t)Bc * 2 * HWc;         // [256][352]
    u16* w2p  = w1p  + (size_t)256 * 352;               // [192][2304]
    u16* wf1p = w2p  + (size_t)192 * 2304;              // [128][128]
    u16* wf2p = wf1p + (size_t)128 * 128;               // [64][1152]
    u16* wop  = wf2p + (size_t)64  * 1152;              // [128][2304]
    u16* flo1_bf = t1_bf;                               // reuse after convc2

    dim3 blk(256);
    auto g1d = [](int n) { return dim3((n + 255) / 256); };

    // ---- precision/layout pre-passes ----
    int ncorr = Bc * 324 * HWc, nflow = Bc * 2 * HWc;
    cvt_bf16<<<g1d(ncorr), blk, 0, stream>>>(corr, corr_bf, ncorr);
    cvt_bf16<<<g1d(nflow), blk, 0, stream>>>(flow, flow_bf, nflow);
    cvt_wpad<<<g1d(256 * 352),  blk, 0, stream>>>(wc1, w1p,  256, 324,  352,  256 * 352);
    cvt_wpad<<<g1d(192 * 2304), blk, 0, stream>>>(wc2, w2p,  192, 2304, 2304, 192 * 2304);
    cvt_wpad<<<g1d(128 * 128),  blk, 0, stream>>>(wf1, wf1p, 128, 98,   128,  128 * 128);
    cvt_wpad<<<g1d(64 * 1152),  blk, 0, stream>>>(wf2, wf2p, 64,  1152, 1152, 64 * 1152);
    cvt_wpad<<<g1d(128 * 2304), blk, 0, stream>>>(wo,  wop,  126, 2304, 2304, 128 * 2304);

    // ---- conv pipeline (stream-ordered) ----
    // convc1: 1x1, 324->256
    conv_wmma<1,1,0,true ><<<dim3(NTOT/128, 4), blk, 0, stream>>>(
        corr_bf, w1p, bc1, t1_bf, 324, 256, 0, 256, 352, 256);
    // convc2: 3x3, 256->192 -> cat channels [0,192)
    conv_wmma<3,3,1,true ><<<dim3(NTOT/128, 3), blk, 0, stream>>>(
        t1_bf, w2p, bc2, cat_bf, 256, 192, 0, 256, 2304, 192);
    // convf1: 7x7, 2->128 (t1 region reused; convc2 already consumed it)
    conv_wmma<7,7,3,true ><<<dim3(NTOT/128, 2), blk, 0, stream>>>(
        flow_bf, wf1p, bf1, flo1_bf, 2, 128, 0, 128, 128, 128);
    // convf2: 3x3, 128->64 -> cat channels [192,256)
    conv_wmma<3,3,1,true ><<<dim3(NTOT/128, 1), blk, 0, stream>>>(
        flo1_bf, wf2p, bf2, cat_bf, 128, 64, 192, 256, 1152, 64);
    // conv: 3x3, 256->126 -> d_out channels [0,126), f32
    conv_wmma<3,3,1,false><<<dim3(NTOT/128, 2), blk, 0, stream>>>(
        cat_bf, wop, bo, out, 256, 126, 0, 128, 2304, 128);
    // concat flow -> d_out channels [126,128)
    copy_flow<<<g1d(nflow), blk, 0, stream>>>(flow, out);
}
